// YaRNMultiHeadAttention_26190710571385
// MI455X (gfx1250) — compile-verified, matrix operand order checked by reading, not run on hardware
//
#include <hip/hip_runtime.h>

// ---------------------------------------------------------------------------
// YaRN multi-head attention forward for MI455X (gfx1250), wave32 + WMMA.
// GEMMs: bf16 WMMA with async-to-LDS double-buffered weight tiles.
// Attention: causal flash with online softmax, all matmuls on WMMA.
// ---------------------------------------------------------------------------

typedef __bf16 bf16_t;
typedef bf16_t v16bf __attribute__((ext_vector_type(16)));
typedef bf16_t v8bf  __attribute__((ext_vector_type(8)));
typedef float  v8f   __attribute__((ext_vector_type(8)));

#define T_SEQ   2048
#define D_MODEL 1024
#define NHEADS  16
#define D_HEAD  64

__device__ __forceinline__ v8f wmma_bf16(v16bf a, v16bf b, v8f c) {
    // (neg_a, A, neg_b, B, c_mod, C, reuse_a, reuse_b)
    return __builtin_amdgcn_wmma_f32_16x16x32_bf16(false, a, false, b,
                                                   (short)0, c, false, false);
}

// Load one 16x32 bf16 fragment row for this lane from a row-major [rows, ld]
// matrix. CDNA5 16-bit A-layout: lanes 0-15 hold K in {kb..kb+7} U
// {kb+16..kb+23}; caller passes kb = k0 + (lane>>4)*8 and row = r0 + (lane&15).
// Both K-groups are 8 contiguous bf16 = one 16-byte load each.
__device__ __forceinline__ v16bf load_frag(const bf16_t* base, int ld, int row, int kb) {
    const v8bf a = *(const v8bf*)(base + (size_t)row * ld + kb);
    const v8bf b = *(const v8bf*)(base + (size_t)row * ld + kb + 16);
    v16bf r;
#pragma unroll
    for (int i = 0; i < 8; ++i) { r[i] = a[i]; r[i + 8] = b[i]; }
    return r;
}

__device__ __forceinline__ float grp16_max(float v) {
#pragma unroll
    for (int m = 1; m < 16; m <<= 1) v = fmaxf(v, __shfl_xor(v, m, 32));
    return v;
}
__device__ __forceinline__ float grp16_sum(float v) {
#pragma unroll
    for (int m = 1; m < 16; m <<= 1) v += __shfl_xor(v, m, 32);
    return v;
}

// ---------------------------------------------------------------------------
__global__ void cvt_f32_to_bf16(const float* __restrict__ in, bf16_t* __restrict__ out, int n) {
    for (int i = blockIdx.x * blockDim.x + threadIdx.x; i < n; i += gridDim.x * blockDim.x)
        out[i] = (bf16_t)in[i];
}

// ---------------------------------------------------------------------------
// Async-copy a 64-row x 32-col bf16 tile of W (row-major [N,K]) into LDS.
// 4 KB total; 128 threads x 16 B x 2 passes.  Tracked by ASYNCcnt.
__device__ __forceinline__ void async_copy_w_tile(const bf16_t* W, int K, int n0, int k0,
                                                  bf16_t* lds_tile, int tid) {
    // Low 32 bits of the flat shared pointer are the LDS byte offset.
    const uint32_t lbase = (uint32_t)(uintptr_t)lds_tile;
#pragma unroll
    for (int p = 0; p < 2; ++p) {
        const int e   = p * 128 + tid;      // 256 x 16B segments (64 rows x 4)
        const int row = e >> 2;
        const int c16 = e & 3;
        const uint64_t ga = (uint64_t)(uintptr_t)(W + (size_t)(n0 + row) * K + k0 + c16 * 8);
        const uint32_t la = lbase + (uint32_t)(row * 64 + c16 * 16);
        asm volatile("global_load_async_to_lds_b128 %0, %1, off"
                     :: "v"(la), "v"(ga) : "memory");
    }
}

// ---------------------------------------------------------------------------
// C[M,N] = A[M,K] @ W[N,K]^T.  Block = 128 threads (4 waves); each wave does a
// 32x64 output tile (8 C fragments); block covers 128x64.  W tiles are staged
// through double-buffered LDS with GLOBAL_LOAD_ASYNC_TO_LDS_B128 and shared by
// all 4 waves; A fragments are loaded straight into registers.
// mode 0: bf16 out row-major.  mode 1: bf16 out transposed (out[n*M+m]).
// mode 2: fp32 out row-major + bias[n].
__global__ void gemm_bf16_wmma(const bf16_t* __restrict__ A, const bf16_t* __restrict__ W,
                               void* __restrict__ outp, const float* __restrict__ bias,
                               int M, int N, int K, int mode) {
    __shared__ __align__(16) bf16_t Ws[2][64 * 32];   // 2 x 4 KB double buffer

    const int tid  = threadIdx.x;
    const int lane = tid & 31;
    const int wave = tid >> 5;
    const int hl   = lane >> 4;
    const int l16  = lane & 15;
    const int m0   = blockIdx.x * 128 + wave * 32;
    const int n0   = blockIdx.y * 64;
    const int nk   = K / 32;

    v8f acc[2][4] = {};

    async_copy_w_tile(W, K, n0, 0, Ws[0], tid);

    for (int ci = 0; ci < nk; ++ci) {
        asm volatile("s_wait_asynccnt 0x0" ::: "memory");
        __syncthreads();                               // W tile ci visible to all waves
        if (ci + 1 < nk)
            async_copy_w_tile(W, K, n0, (ci + 1) * 32, Ws[(ci + 1) & 1], tid);

        const int kb = ci * 32 + hl * 8;
        const v16bf a0 = load_frag(A, K, m0 +      l16, kb);
        const v16bf a1 = load_frag(A, K, m0 + 16 + l16, kb);
        const bf16_t* wt = Ws[ci & 1];
#pragma unroll
        for (int j = 0; j < 4; ++j) {
            const v16bf bf = load_frag(wt, 32, j * 16 + l16, hl * 8);
            acc[0][j] = wmma_bf16(a0, bf, acc[0][j]);
            acc[1][j] = wmma_bf16(a1, bf, acc[1][j]);
        }
        __syncthreads();                               // done reading buffer ci&1
    }

#pragma unroll
    for (int a = 0; a < 2; ++a) {
#pragma unroll
        for (int r = 0; r < 8; ++r) {
            const int m = m0 + a * 16 + r + hl * 8;
#pragma unroll
            for (int j = 0; j < 4; ++j) {
                const int n = n0 + j * 16 + l16;
                const float v = acc[a][j][r];
                if (mode == 2) {
                    ((float*)outp)[(size_t)m * N + n] = v + bias[n];
                } else if (mode == 1) {
                    ((bf16_t*)outp)[(size_t)n * M + m] = (bf16_t)v;
                } else {
                    ((bf16_t*)outp)[(size_t)m * N + n] = (bf16_t)v;
                }
            }
        }
    }
}

// ---------------------------------------------------------------------------
// In-place RoPE on bf16 [T, D] where D = NHEADS * D_HEAD; rotate_half pairs
// (i, i+32) within each head.  One thread per pair.
__global__ void rope_bf16(bf16_t* __restrict__ X, const int* __restrict__ pos) {
    const int idx = blockIdx.x * blockDim.x + threadIdx.x;
    if (idx >= T_SEQ * NHEADS * 32) return;
    const int i = idx & 31;
    const int h = (idx >> 5) & 15;
    const int t = idx >> 9;
    const float p   = (float)pos[t];
    const float inv = __expf(-(float)i * 0.28782313662f);  // ln(10000)/32
    float s, c;
    __sincosf(p * inv, &s, &c);
    bf16_t* row = X + (size_t)t * D_MODEL + h * D_HEAD;
    const float x1 = (float)row[i];
    const float x2 = (float)row[i + 32];
    row[i]      = (bf16_t)(x1 * c - x2 * s);
    row[i + 32] = (bf16_t)(x2 * c + x1 * s);
}

// ---------------------------------------------------------------------------
// Causal flash attention.  grid = (NHEADS, T/64), block = 128 (4 waves).
// Each wave owns 16 query rows; streams 32-key blocks with online softmax.
// Qp, Kp: bf16 [T, D] (RoPE'd).  Vt: bf16 [D, T] (per-head transposed V).
// Ctx: bf16 [T, D] output (pre out-projection).
// NOTE: waves have different causal trip counts -> no block barriers inside
// the loop; per-wave LDS ordering (DS ops are in-order) covers the P relayout.
__global__ void attn_fwd(const bf16_t* __restrict__ Qp, const bf16_t* __restrict__ Kp,
                         const bf16_t* __restrict__ Vt, bf16_t* __restrict__ Ctx) {
    const int lane = threadIdx.x & 31;
    const int wave = threadIdx.x >> 5;
    const int h    = blockIdx.x;
    const int q0   = blockIdx.y * 64 + wave * 16;
    const int hl   = lane >> 4;
    const int l16  = lane & 15;

    // Per-wave staging tile: P (16x32) in A-fragment-loadable row-major form.
    __shared__ __align__(16) bf16_t pbuf[4][16 * 32];
    bf16_t* pb = pbuf[wave];

    const bf16_t* Qb = Qp + h * D_HEAD;
    const bf16_t* Kb = Kp + h * D_HEAD;
    const bf16_t* Vb = Vt + (size_t)h * D_HEAD * T_SEQ;   // rows = d, cols = t

    // Q fragments for this wave's 16 rows, d_k split into two K=32 chunks.
    const v16bf qa0 = load_frag(Qb, D_MODEL, q0 + l16,      hl * 8);
    const v16bf qa1 = load_frag(Qb, D_MODEL, q0 + l16, 32 + hl * 8);

    v8f o0 = {}, o1 = {}, o2 = {}, o3 = {};
    float mrow[8], lrow[8];
#pragma unroll
    for (int r = 0; r < 8; ++r) { mrow[r] = -__builtin_inff(); lrow[r] = 0.0f; }

    const int qhi = q0 + 15;
    for (int nb = 0; nb <= qhi; nb += 32) {
        // ---- scores: S(16x32) = Q(16x64) @ K_blk(32x64)^T, 4 WMMAs ----
        v8f s0 = {}, s1 = {};
        {
            v16bf kf;
            kf = load_frag(Kb, D_MODEL, nb +      l16,      hl * 8); s0 = wmma_bf16(qa0, kf, s0);
            kf = load_frag(Kb, D_MODEL, nb +      l16, 32 + hl * 8); s0 = wmma_bf16(qa1, kf, s0);
            kf = load_frag(Kb, D_MODEL, nb + 16 + l16,      hl * 8); s1 = wmma_bf16(qa0, kf, s1);
            kf = load_frag(Kb, D_MODEL, nb + 16 + l16, 32 + hl * 8); s1 = wmma_bf16(qa1, kf, s1);
        }
        // ---- online softmax per row (row = C-frag element across 16 lanes) ----
#pragma unroll
        for (int r = 0; r < 8; ++r) {
            const int q = q0 + r + hl * 8;
            float a = s0[r] * 0.125f;                 // 1/sqrt(64)
            float b = s1[r] * 0.125f;
            if (nb +      l16 > q) a = -__builtin_inff();
            if (nb + 16 + l16 > q) b = -__builtin_inff();
            const float bm   = grp16_max(fmaxf(a, b));
            const float mnew = fmaxf(mrow[r], bm);
            const float sc   = __expf(mrow[r] - mnew);
            const float p0   = __expf(a - mnew);
            const float p1   = __expf(b - mnew);
            lrow[r] = lrow[r] * sc + grp16_sum(p0 + p1);
            mrow[r] = mnew;
            o0[r] *= sc; o1[r] *= sc; o2[r] *= sc; o3[r] *= sc;
            // C-layout -> row-major LDS tile (A-layout relayout for P*V).
            pb[(r + hl * 8) * 32 +      l16] = (bf16_t)p0;
            pb[(r + hl * 8) * 32 + 16 + l16] = (bf16_t)p1;
        }
        // Same-wave LDS ops are in-order: safe to read back without a barrier.
        const v16bf pa = load_frag(pb, 32, l16, hl * 8);
        // ---- O += P(16x32) @ V_blk(32x64), 4 WMMAs (Vt rows are contiguous) ----
        v16bf vf;
        vf = load_frag(Vb, T_SEQ,      l16, nb + hl * 8); o0 = wmma_bf16(pa, vf, o0);
        vf = load_frag(Vb, T_SEQ, 16 + l16, nb + hl * 8); o1 = wmma_bf16(pa, vf, o1);
        vf = load_frag(Vb, T_SEQ, 32 + l16, nb + hl * 8); o2 = wmma_bf16(pa, vf, o2);
        vf = load_frag(Vb, T_SEQ, 48 + l16, nb + hl * 8); o3 = wmma_bf16(pa, vf, o3);
    }

#pragma unroll
    for (int r = 0; r < 8; ++r) {
        const float inv = 1.0f / lrow[r];
        bf16_t* out = Ctx + (size_t)(q0 + r + hl * 8) * D_MODEL + h * D_HEAD;
        out[     l16] = (bf16_t)(o0[r] * inv);
        out[16 + l16] = (bf16_t)(o1[r] * inv);
        out[32 + l16] = (bf16_t)(o2[r] * inv);
        out[48 + l16] = (bf16_t)(o3[r] * inv);
    }
}

// ---------------------------------------------------------------------------
extern "C" void kernel_launch(void* const* d_in, const int* in_sizes, int n_in,
                              void* d_out, int out_size, void* d_ws, size_t ws_size,
                              hipStream_t stream) {
    const float* query = (const float*)d_in[0];
    const float* key   = (const float*)d_in[1];
    const float* value = (const float*)d_in[2];
    const int*   pos   = (const int*)d_in[3];
    const float* wq    = (const float*)d_in[4];
    const float* wk    = (const float*)d_in[5];
    const float* wv    = (const float*)d_in[6];
    const float* wo_w  = (const float*)d_in[7];
    const float* wo_b  = (const float*)d_in[8];
    float* out = (float*)d_out;

    const size_t nTD = (size_t)T_SEQ * D_MODEL;     // 2M elems
    const size_t nDD = (size_t)D_MODEL * D_MODEL;   // 1M elems

    uint8_t* w = (uint8_t*)d_ws;
    auto carve = [&](size_t bytes) -> void* {
        void* p = (void*)w;
        w += (bytes + 255) & ~(size_t)255;
        return p;
    };
    bf16_t* qb  = (bf16_t*)carve(nTD * 2);
    bf16_t* kb  = (bf16_t*)carve(nTD * 2);
    bf16_t* vb  = (bf16_t*)carve(nTD * 2);
    bf16_t* wqb = (bf16_t*)carve(nDD * 2);
    bf16_t* wkb = (bf16_t*)carve(nDD * 2);
    bf16_t* wvb = (bf16_t*)carve(nDD * 2);
    bf16_t* wob = (bf16_t*)carve(nDD * 2);
    bf16_t* Qp  = (bf16_t*)carve(nTD * 2);
    bf16_t* Kp  = (bf16_t*)carve(nTD * 2);
    bf16_t* Vt  = (bf16_t*)carve(nTD * 2);          // per-head transposed V: [D, T]
    bf16_t* Ctx = (bf16_t*)carve(nTD * 2);

    auto cvt = [&](const float* src, bf16_t* dst, size_t n) {
        int blocks = (int)((n + 255) / 256);
        if (blocks > 4096) blocks = 4096;
        cvt_f32_to_bf16<<<blocks, 256, 0, stream>>>(src, dst, (int)n);
    };
    cvt(query, qb, nTD);
    cvt(key,   kb, nTD);
    cvt(value, vb, nTD);
    cvt(wq, wqb, nDD);
    cvt(wk, wkb, nDD);
    cvt(wv, wvb, nDD);
    cvt(wo_w, wob, nDD);

    const dim3 gemm_grid(T_SEQ / 128, D_MODEL / 64);   // 16 x 16 blocks of 128 thr
    gemm_bf16_wmma<<<gemm_grid, 128, 0, stream>>>(qb, wqb, Qp, nullptr,
                                                  T_SEQ, D_MODEL, D_MODEL, 0);
    gemm_bf16_wmma<<<gemm_grid, 128, 0, stream>>>(kb, wkb, Kp, nullptr,
                                                  T_SEQ, D_MODEL, D_MODEL, 0);
    gemm_bf16_wmma<<<gemm_grid, 128, 0, stream>>>(vb, wvb, Vt, nullptr,
                                                  T_SEQ, D_MODEL, D_MODEL, 1);

    const int rope_threads = T_SEQ * NHEADS * 32;
    rope_bf16<<<(rope_threads + 255) / 256, 256, 0, stream>>>(Qp, pos);
    rope_bf16<<<(rope_threads + 255) / 256, 256, 0, stream>>>(Kp, pos);

    attn_fwd<<<dim3(NHEADS, T_SEQ / 64), 128, 0, stream>>>(Qp, Kp, Vt, Ctx);

    gemm_bf16_wmma<<<gemm_grid, 128, 0, stream>>>(Ctx, wob, out, wo_b,
                                                  T_SEQ, D_MODEL, D_MODEL, 2);
}
